// SplitModel_39651138076861
// MI455X (gfx1250) — matrix operand order repaired and checked by reading, not compiled, not run on hardware
//
#include <hip/hip_runtime.h>
#include <hip/hip_bf16.h>
#include <math.h>

typedef __attribute__((ext_vector_type(16))) _Float16 v16h;
typedef __attribute__((ext_vector_type(8)))  _Float16 v8h;
typedef __attribute__((ext_vector_type(8)))  float    v8f;

#define NNODES 16384
#define KPRED  8
#define DEMB   128
#define HDIM   256
#define TWOD   (2*DEMB)               // 256 = GEMM K dim
#define NEDGE  (NNODES*KPRED)         // 131072
#define MB     64                     // edges (M rows) per block
#define ALS_STRIDE 272                // 256 + 16 halves pad (bank-conflict avoidance)

// ---------------------------------------------------------------------------
// prep: W1 (f32, [256(k) x 256(n)]) -> W1h f16 column-major: W1h[n*256 + k]
//       also zero dists[0] (the prepended zero row of the distance vector)
// ---------------------------------------------------------------------------
__global__ void prep_kernel(const float* __restrict__ W1,
                            _Float16* __restrict__ W1h,
                            float* __restrict__ dists) {
    const int n = blockIdx.x;    // hidden unit 0..255
    const int k = threadIdx.x;   // input dim  0..255
    W1h[n * TWOD + k] = (_Float16)W1[k * HDIM + n];
    if (n == 0 && k == 0) dists[0] = 0.0f;
}

// ---------------------------------------------------------------------------
// edge MLP: d = softplus(tanh(A@W1 + b1) @ W2 + b2), A row = [emb_pred ++ emb_end]
//
// 64 edges per block, 512 threads = 16 waves.  Wave w owns N-tile w (16 hidden
// cols): its 8 B-fragments (16x16x32 chunks of W1) are loaded ONCE into
// registers and reused across 4 M-tiles -> 4x less L2 B-traffic than
// streaming W1 per 16-edge tile (256 MB vs 1 GB for the whole problem).
// Epilogue reduces tanh(c+b1) times W2 across columns with a shfl_xor
// butterfly before touching LDS (16 atomics/wave/M-tile instead of 256).
// ---------------------------------------------------------------------------
__global__ __launch_bounds__(512) void edge_mlp_kernel(
    const float* __restrict__ embs,      // (N+1, 128) f32
    const int*   __restrict__ edges,     // (E, 2) int32: [pred, end]
    const _Float16* __restrict__ W1h,    // (256 n, 256 k) f16
    const float* __restrict__ b1,        // (256,)
    const float* __restrict__ W2,        // (256,1)
    const float* __restrict__ b2,        // (1,)
    float* __restrict__ dists)           // (E+1,)
{
    __shared__ _Float16 Als[MB * ALS_STRIDE];
    __shared__ float score[MB];

    const int tid = threadIdx.x;
    const int m0  = blockIdx.x * MB;     // first edge of this block

    if (tid < MB) score[tid] = 0.0f;

    // ---- stage A panel (64 edges x 256 feats) into LDS, f32 -> f16 ----
    // 64 rows x 16 segments = 1024 tasks over 512 threads.
    #pragma unroll
    for (int task = tid; task < MB * 16; task += 512) {
        const int row  = task >> 4;                // edge row 0..63
        const int seg  = task & 15;                // 16-col segment
        const int e    = m0 + row;
        const int node = (seg < 8) ? edges[2 * e] : edges[2 * e + 1];
        const int segc = (seg & 7) * 16;           // col within the 128-d emb
        const float4* src4 = (const float4*)(embs + (size_t)node * DEMB + segc);
        _Float16* dst = &Als[row * ALS_STRIDE + (seg < 8 ? 0 : DEMB) + segc];
        #pragma unroll
        for (int j = 0; j < 4; ++j) {
            float4 v = src4[j];
            dst[4*j+0] = (_Float16)v.x;
            dst[4*j+1] = (_Float16)v.y;
            dst[4*j+2] = (_Float16)v.z;
            dst[4*j+3] = (_Float16)v.w;
        }
    }
    __syncthreads();

    const int wave  = tid >> 5;          // 0..15  == N-tile index
    const int lane  = tid & 31;
    const int lcol  = lane & 15;
    const int lhalf = lane >> 4;         // 0 or 1
    const int col   = wave * 16 + lcol;  // hidden col (B/C layout: lane -> col N)

    // ---- load this wave's 8 B-fragments once (register-resident W1 tile) ----
    const _Float16* bcol = W1h + (size_t)col * TWOD;
    v16h bfrag[8];
    #pragma unroll
    for (int kk = 0; kk < 8; ++kk) {
        v8h blo = *(const v8h*)(bcol + kk * 32 + lhalf * 8);
        v8h bhi = *(const v8h*)(bcol + kk * 32 + lhalf * 8 + 16);
        #pragma unroll
        for (int j = 0; j < 8; ++j) { bfrag[kk][j] = blo[j]; bfrag[kk][8 + j] = bhi[j]; }
    }

    const float b1v = b1[col];
    const float w2v = W2[col];

    // ---- 4 M-tiles of 16 edges, B fragments reused from registers ----
    #pragma unroll
    for (int mt = 0; mt < 4; ++mt) {
        const _Float16* arow = &Als[(mt * 16 + lcol) * ALS_STRIDE]; // A: lane -> row M

        v8f c = {};
        #pragma unroll
        for (int kk = 0; kk < 8; ++kk) {
            const int kb = kk * 32;
            // A fragment: ISA 16-bit A 16x32 layout -> two contiguous 8-half chunks
            v8h alo = *(const v8h*)(arow + kb + lhalf * 8);
            v8h ahi = *(const v8h*)(arow + kb + lhalf * 8 + 16);
            v16h a;
            #pragma unroll
            for (int j = 0; j < 8; ++j) { a[j] = alo[j]; a[8 + j] = ahi[j]; }
            c = __builtin_amdgcn_wmma_f32_16x16x32_f16(
                    false, a, false, bfrag[kk], (short)0, c, false, false);
        }

        // epilogue: h = tanh(c + b1[col]); score[row] += sum_cols h * W2[col]
        #pragma unroll
        for (int r = 0; r < 8; ++r) {
            float contrib = tanhf(c[r] + b1v) * w2v;
            // reduce across the 16 columns of this half-wave
            contrib += __shfl_xor(contrib, 1, 32);
            contrib += __shfl_xor(contrib, 2, 32);
            contrib += __shfl_xor(contrib, 4, 32);
            contrib += __shfl_xor(contrib, 8, 32);
            if (lcol == 0) {
                const int m = mt * 16 + r + lhalf * 8;  // C layout: VGPR r -> row
                atomicAdd(&score[m], contrib);
            }
        }
    }
    __syncthreads();

    if (tid < MB) {
        const float x  = score[tid] + b2[0];
        // numerically stable softplus = log1p(exp(x))
        const float sp = fmaxf(x, 0.0f) + log1pf(expf(-fabsf(x)));
        dists[m0 + tid + 1] = sp;
    }
}

// ---------------------------------------------------------------------------
// serial forward scan (log-semiring band-8 recurrence) + gold path sum.
// One wave: lanes 0..7 handle the K predecessors; register ring of last 8
// esum values indexed by (node & 7); butterfly shuffles do the 8-lane LSE.
// gold edge (i-1 -> i) == lattice edge (i, k=0), so gold = sum of lane-0 d.
// ---------------------------------------------------------------------------
__global__ void scan_kernel(const float* __restrict__ dists,
                            float* __restrict__ out)
{
    const int lane = threadIdx.x;        // 32 lanes
    float ring = 0.0f;                   // lane j: esum of latest node == j (mod 8)
    float gold = 0.0f;
    float lse  = 0.0f;
    const float NEG = -__builtin_inff();

    for (int i = 1; i <= NNODES; ++i) {
        const int p = i - 1 - lane;                    // predecessor node
        float dval = 0.0f;
        if (lane < 8) dval = dists[(i - 1) * KPRED + lane + 1];
        float es = __shfl(ring, p & 7, 32);
        if (p == 0) es = 0.0f;                         // esum[0] = 0
        const float val = (lane < 8 && p >= 0) ? (es - dval) : NEG;

        float mx = val;
        mx = fmaxf(mx, __shfl_xor(mx, 1, 32));
        mx = fmaxf(mx, __shfl_xor(mx, 2, 32));
        mx = fmaxf(mx, __shfl_xor(mx, 4, 32));
        float s = expf(val - mx);                      // -inf -> 0
        s += __shfl_xor(s, 1, 32);
        s += __shfl_xor(s, 2, 32);
        s += __shfl_xor(s, 4, 32);
        lse = mx + logf(s);

        if (lane == (i & 7)) ring = lse;
        if (lane == 0) gold += dval;                   // k=0 edge == gold edge
    }
    if (lane == 0) out[0] = gold + lse;                // (gold + fwd).mean(), 1 elem
}

// ---------------------------------------------------------------------------
extern "C" void kernel_launch(void* const* d_in, const int* in_sizes, int n_in,
                              void* d_out, int out_size, void* d_ws, size_t ws_size,
                              hipStream_t stream) {
    (void)in_sizes; (void)n_in; (void)out_size; (void)ws_size;
    const float* embs  = (const float*)d_in[0];   // words_embs (N+1, 128)
    const float* W1    = (const float*)d_in[1];   // (256, 256)
    const float* b1    = (const float*)d_in[2];   // (256,)
    const float* W2    = (const float*)d_in[3];   // (256, 1)
    const float* b2    = (const float*)d_in[4];   // (1,)
    const int*   edges = (const int*)d_in[6];     // (E, 2)

    // workspace layout
    float* dists = (float*)d_ws;                                  // (E+1) f32
    size_t off = (((size_t)(NEDGE + 1) * sizeof(float)) + 255) & ~(size_t)255;
    _Float16* W1h = (_Float16*)((char*)d_ws + off);               // 256*256 f16

    prep_kernel<<<HDIM, TWOD, 0, stream>>>(W1, W1h, dists);
    edge_mlp_kernel<<<NEDGE / MB, 512, 0, stream>>>(
        embs, edges, W1h, b1, W2, b2, dists);
    scan_kernel<<<1, 32, 0, stream>>>(dists, (float*)d_out);
}
